// VGG9_11424613008091
// MI455X (gfx1250) — matrix operand order repaired and checked by reading, not compiled, not run on hardware
//
#include <hip/hip_runtime.h>
#include <hip/hip_fp16.h>

typedef __attribute__((ext_vector_type(16))) _Float16 v16h;
typedef __attribute__((ext_vector_type(8)))  _Float16 v8h;
typedef __attribute__((ext_vector_type(8)))  float    v8f;
typedef __attribute__((ext_vector_type(4)))  unsigned v4u;
typedef __attribute__((ext_vector_type(8)))  int      v8i;
typedef __attribute__((ext_vector_type(4)))  int      v4i;

#define N_    32
#define C_    256
#define H_    56
#define W_    56
#define HP    58          // padded rows (1 top + 1 bottom)
#define WPAD  66          // padded cols (1 left + 9 right)
#define NSTEP 72          // 9 taps * 8 ci-blocks of 32
#define WSWZ  589824      // 9*8*16*32*16 swizzled weight halfs per sign
#define XPITCH 40         // LDS halfs per pixel (32 ci + 4DW TDM pad) = 80B
#define PXT   128         // px per workgroup tile (2 output rows of 64)

// ---- ADC output quantization: linspace(0, 8, 73728) round-to-nearest, saturate ----
__device__ __forceinline__ float quant_out(float v) {
    const float step    = 8.0f / 73727.0f;
    const float invstep = 73727.0f / 8.0f;
    float idx = floorf(v * invstep + 0.5f);
    idx = fminf(fmaxf(idx, 0.0f), 73727.0f);
    return idx * step;
}

// ============================================================================
// Prep 1: DAC-quantize x (256 levels over [0,8]) -> fp16, channels-last padded
// layout [n][58][66][256] with zero halo. Channels-last makes every WMMA
// B-fragment (32 consecutive ci at one pixel) contiguous, so conv tap shifts
// become 512B-aligned pointer offsets and the hot loop needs no transpose.
// ============================================================================
__global__ __launch_bounds__(256) void quant_pad_nhwc_kernel(
    const float* __restrict__ x, _Float16* __restrict__ xq) {
    const int blk = blockIdx.x;            // n*58 + rp
    const int rp  = blk % HP;
    const int n   = blk / HP;
    const int ci  = threadIdx.x;           // 0..255
    const bool rowok = (rp >= 1 && rp <= H_);
    const size_t src_row = ((size_t)(n * C_ + ci) * H_ + (size_t)(rp - 1)) * W_;
    const size_t dst_row = ((size_t)blk * WPAD) * C_ + ci;
    for (int w = 0; w < WPAD; ++w) {
        float v = 0.0f;
        if (rowok && w >= 1 && w <= W_) {
            float xx = x[src_row + (w - 1)];
            float idx = floorf(xx * (255.0f / 8.0f) + 0.5f);
            idx = fminf(fmaxf(idx, 0.0f), 255.0f);
            v = idx * (8.0f / 255.0f);
        }
        xq[dst_row + (size_t)w * C_] = (_Float16)v;
    }
}

// ============================================================================
// Prep 2: split weights into pos/neg conductances, pre-swizzled into exact
// WMMA A-fragment order (16-bit A 16x32, ISA 7.12.2):
//   flat = [tap(9)][ciblk(8)][co_subtile(16)][lane(32)][half(16)]
//   lane l -> co = sub*16 + (l&15);  half h -> K = (l<16?0:8) + (h<8? h : h+8)
// so the hot loop async-copies contiguous 16B/lane chunks straight to LDS.
// ============================================================================
__global__ void weight_swizzle_kernel(const float* __restrict__ w,
                                      _Float16* __restrict__ wpos,
                                      _Float16* __restrict__ wneg) {
    size_t i = (size_t)blockIdx.x * blockDim.x + threadIdx.x;
    if (i >= (size_t)WSWZ) return;
    int hh = (int)(i & 15);
    int l  = (int)((i >> 4) & 31);
    int sg = (int)((i >> 9) & 15);
    int cb = (int)((i >> 13) & 7);
    int t  = (int)(i >> 16);
    int k  = ((l & 16) ? 8 : 0) + ((hh < 8) ? hh : hh + 8);
    int ci = cb * 32 + k;
    int co = sg * 16 + (l & 15);
    int kh = t / 3, kw = t - kh * 3;
    float wv = w[(((size_t)co * C_ + ci) * 3 + kh) * 3 + kw];
    wpos[i] = (_Float16)fmaxf(wv, 0.0f);
    wneg[i] = (_Float16)fmaxf(-wv, 0.0f);
}

// ============================================================================
// Main conv kernel: implicit GEMM on v_wmma_f32_16x16x32_f16.
// Grid: (h-pair=28, n=32, cog=4).  Block: 256 threads = 8 waves.
// Tile: 64 co x 128 px (TWO output rows); K loop = 72 steps (9 taps x 8 cb).
// Per step:  X tile (32ci x 64px x 2rows) -> LDS via ONE 3D tensor_load_to_lds
// (TDM; dim0 stride 256, dim1 stride 66*256, pad 16DW/4DW -> uniform 80B LDS
// pitch), issued by wave 0 via a scalar branch; weight fragments (pos+neg,
// 8KB) -> LDS via global_load_async_to_lds_b128.
// Compute: per wave 12x ds_load_b128 fragments feeding 8 WMMAs per step
// (4 px-subtiles x pos/neg). Double-buffered with s_wait_tensorcnt /
// s_wait_asynccnt + barrier.
// ============================================================================
__global__ __launch_bounds__(256) void conv_wmma_kernel(
    const _Float16* __restrict__ xq,
    const _Float16* __restrict__ wpos,
    const _Float16* __restrict__ wneg,
    float* __restrict__ out) {

    __shared__ _Float16 shX[2][PXT * XPITCH]; // [px][ci] + TDM pad, 80B pitch
    __shared__ _Float16 shW[2][4096];         // [0..2047]=pos frags, [2048..]=neg

    const int tid  = threadIdx.x;
    const int h0   = blockIdx.x * 2;          // first of two output rows
    const int n    = blockIdx.y;
    const int cog  = blockIdx.z;              // co group of 64
    const int wave = tid >> 5;
    const int lane = tid & 31;
    const int sl   = wave >> 1;               // co subtile 0..3 within group
    const int pH   = wave & 1;                // px half (64 px)
    const bool issuer = (__builtin_amdgcn_readfirstlane(tid) == 0); // scalar: wave 0

    auto stage = [&](int kk, int buf) {
        const int tap = kk >> 3, cb = kk & 7;
        const int kh = tap / 3, kw = tap - kh * 3;

        // ---- weights: 4KB pos + 4KB neg; one async b128 chunk per thread ----
        {
            size_t off = (size_t)kk * 8192 + (size_t)cog * 2048 + (size_t)tid * 8;
            unsigned lp = (unsigned)(unsigned long long)&shW[buf][tid * 8];
            unsigned long long gp = (unsigned long long)(wpos + off);
            asm volatile("global_load_async_to_lds_b128 %0, %1, off"
                         :: "v"(lp), "v"(gp) : "memory");
            unsigned ln = (unsigned)(unsigned long long)&shW[buf][2048 + tid * 8];
            unsigned long long gn = (unsigned long long)(wneg + off);
            asm volatile("global_load_async_to_lds_b128 %0, %1, off"
                         :: "v"(ln), "v"(gn) : "memory");
        }

        // ---- X tile via one 3D TDM: 32(ci) x 64(px) x 2(rows) ----
        if (issuer) {
            const size_t off_h =
                (((size_t)(n * HP + (h0 + kh))) * WPAD + kw) * C_ + (size_t)cb * 32;
            const unsigned long long ga = (unsigned long long)(xq + off_h);
            const unsigned lds = (unsigned)(unsigned long long)&shX[buf][0];

            v4u g0;
            g0.x = 1u;                                   // count=1, user mode
            g0.y = lds;                                  // lds_addr
            g0.z = (unsigned)(ga & 0xffffffffull);       // global_addr[31:0]
            g0.w = (unsigned)((ga >> 32) & 0x1ffffffull) // global_addr[56:32]
                 | (2u << 30);                           // type=2 ("image")

            v8i g1;
            g1[0] = (1 << 16)          // data_size = 2 bytes
                  | (1 << 20)          // pad_enable
                  | (3 << 22)          // pad_interval: every 16 DWORDs (64B)
                  | (3 << 25);         // pad_amount: 4 DWORDs (16B)
            g1[1] = (32 << 16);        // tensor_dim0 = 32 (lo16)
            g1[2] = (64 << 16);        // tensor_dim0 hi=0 | tensor_dim1=64 lo16
            g1[3] = (32 << 16);        // tensor_dim1 hi=0 | tile_dim0 = 32
            g1[4] = 64 | (2 << 16);    // tile_dim1 = 64, tile_dim2 = 2 (rows)
            g1[5] = 256;               // tensor_dim0_stride = 256 elems (1 px)
            g1[6] = (int)((WPAD * C_) << 16); // tensor_dim1_stride lo16 (=16896)
            g1[7] = 0;                 // tensor_dim1_stride hi bits
            v4i g2 = {2, 0, 0, 0};     // tensor_dim2 = 2; dim3/tile_dim3 unused
            v4i g3 = {0, 0, 0, 0};
            v8i g4 = {0, 0, 0, 0, 0, 0, 0, 0};
            __builtin_amdgcn_tensor_load_to_lds(g0, g1, g2, g3, g4, 0);
        }
    };

    v8f accP[4] = {v8f{}, v8f{}, v8f{}, v8f{}};
    v8f accN[4] = {v8f{}, v8f{}, v8f{}, v8f{}};

    stage(0, 0);
    for (int kk = 0; kk < NSTEP; ++kk) {
        const int buf = kk & 1;
        __builtin_amdgcn_s_wait_tensorcnt(0);           // no-op for waves 1..7
        asm volatile("s_wait_asynccnt 0x0" ::: "memory");
        __syncthreads();
        if (kk + 1 < NSTEP) stage(kk + 1, buf ^ 1);

        union Frag { v16h v; v8h h2[2]; };
        Frag aP, aN, b[4];
        const _Float16* wb = &shW[buf][0];
        const int abase = (sl * 32 + lane) * 16;
        aP.h2[0] = *(const v8h*)(wb + abase);
        aP.h2[1] = *(const v8h*)(wb + abase + 8);
        aN.h2[0] = *(const v8h*)(wb + 2048 + abase);
        aN.h2[1] = *(const v8h*)(wb + 2048 + abase + 8);

        const _Float16* xb = &shX[buf][0];
        const int koff = (lane >> 4) * 8;        // K 0-7,16-23 / 8-15,24-31 per B layout
        #pragma unroll
        for (int ps = 0; ps < 4; ++ps) {
            const int px0 = pH * 64 + ps * 16 + (lane & 15);
            b[ps].h2[0] = *(const v8h*)(xb + px0 * XPITCH + koff);
            b[ps].h2[1] = *(const v8h*)(xb + px0 * XPITCH + koff + 16);
        }
        #pragma unroll
        for (int ps = 0; ps < 4; ++ps) {
            accP[ps] = __builtin_amdgcn_wmma_f32_16x16x32_f16(
                false, aP.v, false, b[ps].v, (short)0, accP[ps], false, false);
            accN[ps] = __builtin_amdgcn_wmma_f32_16x16x32_f16(
                false, aN.v, false, b[ps].v, (short)0, accN[ps], false, false);
        }
    }

    // ---- epilogue: ADC requant both polarities, store qpos - qneg ----
    // C/D layout: VGPR r, lanes0-15 -> M=r, lanes16-31 -> M=r+8; N = lane&15.
    const int coB = cog * 64 + sl * 16 + (lane >> 4) * 8;
    #pragma unroll
    for (int ps = 0; ps < 4; ++ps) {
        const int pxp = pH * 64 + ps * 16 + (lane & 15);
        const int row = pxp >> 6;          // 0 or 1 -> output row h0+row
        const int w   = pxp & 63;
        if (w < W_) {
            #pragma unroll
            for (int r = 0; r < 8; ++r) {
                const int co = coB + r;
                size_t o = (((size_t)n * C_ + co) * H_ + (h0 + row)) * W_ + w;
                out[o] = quant_out(accP[ps][r]) - quant_out(accN[ps][r]);
            }
        }
    }
}

// ============================================================================
extern "C" void kernel_launch(void* const* d_in, const int* in_sizes, int n_in,
                              void* d_out, int out_size, void* d_ws, size_t ws_size,
                              hipStream_t stream) {
    (void)in_sizes; (void)n_in; (void)out_size; (void)ws_size;
    const float* x = (const float*)d_in[0];
    const float* w = (const float*)d_in[1];
    float* out = (float*)d_out;

    _Float16* xq   = (_Float16*)d_ws;                        // 62.7 MB NHWC fp16
    _Float16* wpos = xq + (size_t)N_ * HP * WPAD * C_;       // 1.18 MB
    _Float16* wneg = wpos + (size_t)WSWZ;                    // 1.18 MB

    quant_pad_nhwc_kernel<<<dim3(N_ * HP), dim3(256), 0, stream>>>(x, xq);
    weight_swizzle_kernel<<<dim3((WSWZ + 255) / 256), dim3(256), 0, stream>>>(w, wpos, wneg);

    dim3 grid(H_ / 2, N_, 4);
    conv_wmma_kernel<<<grid, dim3(256), 0, stream>>>(xq, wpos, wneg, out);
}
